// Seq2Seq_17738214932853
// MI455X (gfx1250) — matrix operand order repaired and checked by reading, compiled, not run on hardware
//
#include <hip/hip_runtime.h>

// ---------------------------------------------------------------------------
// Seq2Seq (2-layer biLSTM encoder + 2-layer attention decoder) for gfx1250.
// GEMMs run on v_wmma_f32_16x16x32_f16 (f16 in, f32 acc); A-panels staged to
// LDS via the Tensor Data Mover (tensor_load_to_lds + s_wait_tensorcnt).
// ---------------------------------------------------------------------------

typedef __attribute__((ext_vector_type(16))) _Float16 v16h;
typedef __attribute__((ext_vector_type(8)))  _Float16 v8h;
typedef __attribute__((ext_vector_type(8)))  float    v8f;
typedef __attribute__((ext_vector_type(4)))  unsigned v4u;
typedef __attribute__((ext_vector_type(8)))  int      v8i;
typedef __attribute__((ext_vector_type(4)))  int      v4i;

constexpr int HID  = 512;    // H
constexpr int GATE = 2048;   // 4H
constexpr int NB   = 128;    // batch
constexpr int LSEQ = 256;    // encoder length
constexpr int TDEC = 10;     // decoder steps
constexpr int VOC  = 27;     // vocab

// ---------------------------------------------------------------------------
// WMMA GEMM:  C[M,Ncols](f32) (+)= A[M,K](f16) * B[Ncols,K](f16)^T (+ bias)
// Optional f16 mirror of the result in Ch. Block = 8 waves, covers 16 rows x
// 128 cols; grid = (Ncols/128, M/16). K in {512,1024}, multiple of 32.
// A panel (16 x K, <=32KB) is DMA'd to LDS by the TDM from wave 0.
// ---------------------------------------------------------------------------
__global__ __launch_bounds__(256) void wmma_gemm(
    const _Float16* __restrict__ A, const _Float16* __restrict__ B,
    float* __restrict__ C, _Float16* __restrict__ Ch,
    const float* __restrict__ bias,
    int K, int Ncols, int accumulate)
{
  __shared__ _Float16 Als[16 * 1024];

  const int lane = threadIdx.x & 31;
  const int wv   = threadIdx.x >> 5;
  const int hl   = lane >> 4;     // lane half: 0 -> lanes 0..15, 1 -> 16..31
  const int mr   = lane & 15;

#if __has_builtin(__builtin_amdgcn_tensor_load_to_lds)
  if (wv == 0) {
    // Tensor DMA descriptor: 2-D tile, 16 rows x K f16 elements, row-major.
    unsigned long long ga =
        (unsigned long long)(size_t)(A + (size_t)blockIdx.y * 16 * K);
    unsigned ldsoff = (unsigned)(size_t)(&Als[0]);   // flat addr low 32 = LDS offset
    v4u g0;
    g0[0] = 1u;                                       // count = 1 (valid D#)
    g0[1] = ldsoff;                                   // lds_addr
    g0[2] = (unsigned)ga;                             // global_addr[31:0]
    g0[3] = (unsigned)(ga >> 32) | (2u << 30);        // global_addr[56:32] | type=2
    v8i g1;
    g1[0] = (int)(1u << 16);                          // data_size = 1 (2 bytes)
    g1[1] = (int)((unsigned)(K & 0xFFFF) << 16);      // tensor_dim0[15:0]
    g1[2] = (int)(16u << 16);                         // tensor_dim1 = 16
    g1[3] = (int)((unsigned)(K & 0xFFFF) << 16);      // tile_dim0 = K
    g1[4] = 16;                                       // tile_dim1 = 16
    g1[5] = K;                                        // tensor_dim0_stride = K
    g1[6] = 0;
    g1[7] = 0;
    v4i gz4; gz4[0] = 0; gz4[1] = 0; gz4[2] = 0; gz4[3] = 0;
    v8i gz8;
    gz8[0] = 0; gz8[1] = 0; gz8[2] = 0; gz8[3] = 0;
    gz8[4] = 0; gz8[5] = 0; gz8[6] = 0; gz8[7] = 0;
    __builtin_amdgcn_tensor_load_to_lds(g0, g1, gz4, gz4, gz8, 0);
    __builtin_amdgcn_s_wait_tensorcnt(0);
  }
#else
  {
    const _Float16* src = A + (size_t)blockIdx.y * 16 * K;
    for (int i = threadIdx.x; i < 16 * K; i += 256) Als[i] = src[i];
  }
#endif
  __syncthreads();

  const int col0 = blockIdx.x * 128 + wv * 16;

  // A fragment (16x32 f16): lanes 0..15 hold row mr, K = {k0..k0+15};
  // lanes 16..31 hold K = {k0+16..31}.
  const _Float16* Ap = Als + mr * K + hl * 16;
  // B fragment (32x16 f16): lane holds column (col0+mr), 16 contiguous K.
  const _Float16* Bp = B + (size_t)(col0 + mr) * K + hl * 16;

  v8f acc = {};
  for (int k0 = 0; k0 < K; k0 += 32) {
    union { v16h v; v8h h[2]; } a;
    a.h[0] = *(const v8h*)(Ap + k0);
    a.h[1] = *(const v8h*)(Ap + k0 + 8);
    v16h b = *(const v16h*)(Bp + k0);
    acc = __builtin_amdgcn_wmma_f32_16x16x32_f16(
        false, a.v, false, b, (short)0, acc, false, false);
  }

  // C/D layout: VGPR r -> row (hl*8 + r), column = lane & 15.
  const int rowBase = blockIdx.y * 16 + hl * 8;
  const int col = col0 + mr;
  #pragma unroll
  for (int r = 0; r < 8; ++r) {
    size_t idx = (size_t)(rowBase + r) * Ncols + col;
    float v = acc[r];
    if (bias) v += bias[col];
    if (accumulate && C) v += C[idx];
    if (C)  C[idx]  = v;
    if (Ch) Ch[idx] = (_Float16)v;
  }
}

// ---------------------------------------------------------------------------
// Encoder layer-0 input projection: z[n,g] = sum_{k<6} x[n,l,k] * Wih[g,k]
// ---------------------------------------------------------------------------
__global__ __launch_bounds__(256) void xproj6(
    float* __restrict__ z, const float* __restrict__ input,
    const float* __restrict__ Wih, int l)
{
  int idx = blockIdx.x * 256 + threadIdx.x;      // n*GATE + g
  int n = idx >> 11, g = idx & 2047;
  const float* xr = input + ((size_t)n * LSEQ + l) * 6;
  const float* w  = Wih + g * 6;
  float s = 0.f;
  #pragma unroll
  for (int k = 0; k < 6; ++k) s += xr[k] * w[k];
  z[idx] = s;
}

// ---------------------------------------------------------------------------
// Fused LSTM pointwise cell (PyTorch gate order i,f,g,o).
// ---------------------------------------------------------------------------
__global__ __launch_bounds__(256) void lstm_cell(
    const float* __restrict__ z,
    const float* __restrict__ bih, const float* __restrict__ bhh,
    float* __restrict__ c,                       // in/out
    float* __restrict__ h32, _Float16* __restrict__ h16,
    _Float16* __restrict__ yout, int ystride)
{
  int idx = blockIdx.x * 256 + threadIdx.x;      // n*HID + j
  int n = idx >> 9, j = idx & 511;
  const float* zr = z + (size_t)n * GATE;
  float zi = zr[j]           + bih[j]           + bhh[j];
  float zf = zr[j + HID]     + bih[j + HID]     + bhh[j + HID];
  float zg = zr[j + 2 * HID] + bih[j + 2 * HID] + bhh[j + 2 * HID];
  float zo = zr[j + 3 * HID] + bih[j + 3 * HID] + bhh[j + 3 * HID];
  float si = 1.f / (1.f + __expf(-zi));
  float sf = 1.f / (1.f + __expf(-zf));
  float so = 1.f / (1.f + __expf(-zo));
  float cn = sf * c[idx] + si * tanhf(zg);
  float hn = so * tanhf(cn);
  c[idx] = cn;
  if (h32) h32[idx] = hn;
  h16[idx] = (_Float16)hn;
  if (yout) yout[(size_t)n * ystride + j] = (_Float16)hn;
}

// Row softmax over 512 cols (one block per row).
__global__ __launch_bounds__(256) void softmax512(
    const float* __restrict__ src, float* __restrict__ dst32,
    _Float16* __restrict__ dst16)
{
  __shared__ float red[256];
  int n = blockIdx.x, t = threadIdx.x;
  const float* s = src + (size_t)n * HID;
  float a = s[t], b = s[t + 256];
  red[t] = fmaxf(a, b); __syncthreads();
  for (int o = 128; o > 0; o >>= 1) { if (t < o) red[t] = fmaxf(red[t], red[t + o]); __syncthreads(); }
  float m = red[0]; __syncthreads();
  float ea = __expf(a - m), eb = __expf(b - m);
  red[t] = ea + eb; __syncthreads();
  for (int o = 128; o > 0; o >>= 1) { if (t < o) red[t] += red[t + o]; __syncthreads(); }
  float inv = 1.f / red[0];
  float ra = ea * inv, rb = eb * inv;
  dst32[(size_t)n * HID + t] = ra;
  dst32[(size_t)n * HID + t + 256] = rb;
  if (dst16) {
    dst16[(size_t)n * HID + t] = (_Float16)ra;
    dst16[(size_t)n * HID + t + 256] = (_Float16)rb;
  }
}

// dst[n, 0:512] = a[n,:], dst[n, 512:1024] = b[n,:]  (f32 -> f16)
__global__ __launch_bounds__(256) void concat2(
    _Float16* __restrict__ dst, const float* __restrict__ a,
    const float* __restrict__ b)
{
  int idx = blockIdx.x * 256 + threadIdx.x;      // n*1024 + cc
  int n = idx >> 10, cc = idx & 1023;
  float v = (cc < HID) ? a[(size_t)n * HID + cc] : b[(size_t)n * HID + cc - HID];
  dst[idx] = (_Float16)v;
}

// Attention: one block per batch row. scores over L=256 (== blockDim), then
// softmax, then ctx[n, :] = attn @ val.
__global__ __launch_bounds__(256) void attn_kernel(
    const float* __restrict__ q,                 // [NB, HID]
    const _Float16* __restrict__ key,            // [(L*NB), HID] time-major
    const _Float16* __restrict__ val,
    float* __restrict__ ctx)                     // [NB, HID]
{
  __shared__ float sc[256];
  __shared__ float red[256];
  int n = blockIdx.x, t = threadIdx.x;
  const float scale = 0.044194173824159216f;     // 1/sqrt(512)
  const float* qr = q + (size_t)n * HID;
  const _Float16* kr = key + ((size_t)t * NB + n) * HID;
  float s = 0.f;
  for (int j = 0; j < HID; ++j) s += qr[j] * (float)kr[j];
  s *= scale;
  red[t] = s; __syncthreads();
  for (int o = 128; o > 0; o >>= 1) { if (t < o) red[t] = fmaxf(red[t], red[t + o]); __syncthreads(); }
  float m = red[0]; __syncthreads();
  float e = __expf(s - m);
  red[t] = e; __syncthreads();
  for (int o = 128; o > 0; o >>= 1) { if (t < o) red[t] += red[t + o]; __syncthreads(); }
  float inv = 1.f / red[0]; __syncthreads();
  sc[t] = e * inv; __syncthreads();
  for (int j = t; j < HID; j += 256) {
    float accum = 0.f;
    for (int l = 0; l < LSEQ; ++l)
      accum += sc[l] * (float)val[((size_t)l * NB + n) * HID + j];
    ctx[(size_t)n * HID + j] = accum;
  }
}

// Decoder input: x[n] = [ emb[tok_n] , ctx[n] ]  (f16, width 1024)
__global__ __launch_bounds__(256) void build_x(
    _Float16* __restrict__ x, const float* __restrict__ emb,
    const int* __restrict__ words, int t, const float* __restrict__ ctx)
{
  int idx = blockIdx.x * 256 + threadIdx.x;      // n*1024 + cc
  int n = idx >> 10, cc = idx & 1023;
  float v;
  if (cc < HID) {
    int tok = (t == 0) ? 0 : words[n * TDEC + (t - 1)];
    v = emb[(size_t)tok * HID + cc];
  } else {
    v = ctx[(size_t)n * HID + cc - HID];
  }
  x[idx] = (_Float16)v;
}

// Final FC (V=27) + softmax; one 32-thread block per batch row.
__global__ __launch_bounds__(32) void fc_softmax(
    const float* __restrict__ h1, const float* __restrict__ fcW,
    const float* __restrict__ fcb, float* __restrict__ out, int t)
{
  __shared__ float lg[32];
  int n = blockIdx.x, v = threadIdx.x;
  float s = -1e30f;
  if (v < VOC) {
    const float* hr = h1 + (size_t)n * HID;
    const float* w  = fcW + (size_t)v * HID;
    float a = fcb[v];
    for (int j = 0; j < HID; ++j) a += hr[j] * w[j];
    s = a;
  }
  lg[v] = s; __syncthreads();
  float m = -1e30f;
  for (int j = 0; j < VOC; ++j) m = fmaxf(m, lg[j]);
  __syncthreads();
  float e = (v < VOC) ? __expf(s - m) : 0.f;
  lg[v] = e; __syncthreads();
  float sum = 0.f;
  for (int j = 0; j < VOC; ++j) sum += lg[j];
  if (v < VOC) out[((size_t)n * TDEC + t) * VOC + v] = e / sum;
}

__global__ __launch_bounds__(256) void cvt_f32_f16(
    _Float16* __restrict__ d, const float* __restrict__ s, int n)
{
  int i = blockIdx.x * 256 + threadIdx.x;
  if (i < n) d[i] = (_Float16)s[i];
}

__global__ __launch_bounds__(256) void fill_zero_f32(float* p, int n)
{
  int i = blockIdx.x * 256 + threadIdx.x;
  if (i < n) p[i] = 0.f;
}

__global__ __launch_bounds__(256) void fill_zero_f16(_Float16* p, int n)
{
  int i = blockIdx.x * 256 + threadIdx.x;
  if (i < n) p[i] = (_Float16)0.f;
}

// ---------------------------------------------------------------------------
extern "C" void kernel_launch(void* const* d_in, const int* in_sizes, int n_in,
                              void* d_out, int out_size, void* d_ws, size_t ws_size,
                              hipStream_t stream)
{
  (void)in_sizes; (void)n_in; (void)out_size; (void)ws_size;

  const float* input = (const float*)d_in[0];
  const int*   words = (const int*)d_in[1];
  // runs: 0=e0f 1=e0b 2=e1f 3=e1b 4=d0 5=d1 ; each: Wih,Whh,bih,bhh
  const float* W[6][4];
  for (int i = 0; i < 6; ++i)
    for (int j = 0; j < 4; ++j)
      W[i][j] = (const float*)d_in[2 + i * 4 + j];
  const float* emb  = (const float*)d_in[26];
  const float* fcW  = (const float*)d_in[27];
  const float* fcb  = (const float*)d_in[28];
  const float* hp1W = (const float*)d_in[29]; const float* hp1b = (const float*)d_in[30];
  const float* hp2W = (const float*)d_in[31]; const float* hp2b = (const float*)d_in[32];
  const float* cp1W = (const float*)d_in[33]; const float* cp1b = (const float*)d_in[34];
  const float* cp2W = (const float*)d_in[35]; const float* cp2b = (const float*)d_in[36];
  const float* kpW  = (const float*)d_in[37]; const float* kpb  = (const float*)d_in[38];
  const float* vpW  = (const float*)d_in[39]; const float* vpb  = (const float*)d_in[40];
  float* out = (float*)d_out;

  size_t off = 0;
  auto alloc = [&](size_t bytes) -> void* {
    void* p = (char*)d_ws + off;
    off = (off + bytes + 255) & ~(size_t)255;
    return p;
  };
  auto cvt = [&](const float* src, int n) -> _Float16* {
    _Float16* d = (_Float16*)alloc((size_t)n * 2);
    cvt_f32_f16<<<(n + 255) / 256, 256, 0, stream>>>(d, src, n);
    return d;
  };

  // ---- f16 weight copies (all GEMM operands) ----
  _Float16* wh16[6];
  for (int i = 0; i < 6; ++i) wh16[i] = cvt(W[i][1], GATE * HID);     // Whh
  _Float16* wi16_e1f = cvt(W[2][0], GATE * 1024);
  _Float16* wi16_e1b = cvt(W[3][0], GATE * 1024);
  _Float16* wi16_d0  = cvt(W[4][0], GATE * 1024);
  _Float16* wi16_d1  = cvt(W[5][0], GATE * HID);
  _Float16* hp1_16 = cvt(hp1W, HID * 1024);
  _Float16* hp2_16 = cvt(hp2W, HID * 1024);
  _Float16* cp1_16 = cvt(cp1W, HID * 1024);
  _Float16* cp2_16 = cvt(cp2W, HID * 1024);
  _Float16* kp_16  = cvt(kpW,  HID * 1024);
  _Float16* vp_16  = cvt(vpW,  HID * 1024);

  // ---- big intermediates (f16) ----
  _Float16* y0  = (_Float16*)alloc((size_t)LSEQ * NB * 1024 * 2);  // layer0 output [L,N,2H]
  _Float16* enc = (_Float16*)alloc((size_t)LSEQ * NB * 1024 * 2);  // layer1 output [L,N,2H]
  _Float16* key = (_Float16*)alloc((size_t)LSEQ * NB * HID * 2);
  _Float16* val = (_Float16*)alloc((size_t)LSEQ * NB * HID * 2);
  float* z = (float*)alloc((size_t)NB * GATE * 4);                 // per-step gate buffer

  // ---- states ----
  _Float16* h16[4]; float* h32[4]; float* c32[4];
  for (int r = 0; r < 4; ++r) {
    h16[r] = (_Float16*)alloc((size_t)NB * HID * 2);
    h32[r] = (float*)alloc((size_t)NB * HID * 4);
    c32[r] = (float*)alloc((size_t)NB * HID * 4);
  }
  float* dh32[2]; _Float16* dh16[2]; float* dc[2];
  for (int r = 0; r < 2; ++r) {
    dh32[r] = (float*)alloc((size_t)NB * HID * 4);
    dh16[r] = (_Float16*)alloc((size_t)NB * HID * 2);
    dc[r]   = (float*)alloc((size_t)NB * HID * 4);
  }
  float* proj_pre = (float*)alloc((size_t)NB * HID * 4);
  _Float16* catbuf = (_Float16*)alloc((size_t)NB * 1024 * 2);
  float* ctx = (float*)alloc((size_t)NB * HID * 4);
  _Float16* xbuf = (_Float16*)alloc((size_t)NB * 1024 * 2);

  const dim3 gGate(GATE / 128, NB / 16);   // 16 x 8 blocks (output [128,2048])
  const dim3 gHid(HID / 128, NB / 16);     // 4 x 8 blocks  (output [128,512])

  auto zero_state = [&](int r) {
    fill_zero_f16<<<(NB * HID + 255) / 256, 256, 0, stream>>>(h16[r], NB * HID);
    fill_zero_f32<<<(NB * HID + 255) / 256, 256, 0, stream>>>(c32[r], NB * HID);
  };

  // ---- encoder layer 0 (K=6 input proj on VALU, recurrence on WMMA) ----
  auto run_l0 = [&](int r, bool rev) {
    zero_state(r);
    for (int s = 0; s < LSEQ; ++s) {
      int l = rev ? (LSEQ - 1 - s) : s;
      xproj6<<<(NB * GATE) / 256, 256, 0, stream>>>(z, input, W[r][0], l);
      wmma_gemm<<<gGate, 256, 0, stream>>>(h16[r], wh16[r], z, nullptr, nullptr,
                                           HID, GATE, 1);
      lstm_cell<<<(NB * HID) / 256, 256, 0, stream>>>(
          z, W[r][2], W[r][3], c32[r], h32[r], h16[r],
          y0 + (size_t)l * NB * 1024 + (rev ? HID : 0), 1024);
    }
  };
  run_l0(0, false);
  run_l0(1, true);

  // ---- encoder layer 1 (input proj + recurrence both WMMA) ----
  auto run_l1 = [&](int r, const _Float16* wi, bool rev) {
    zero_state(r);
    for (int s = 0; s < LSEQ; ++s) {
      int l = rev ? (LSEQ - 1 - s) : s;
      wmma_gemm<<<gGate, 256, 0, stream>>>(y0 + (size_t)l * NB * 1024, wi, z,
                                           nullptr, nullptr, 1024, GATE, 0);
      wmma_gemm<<<gGate, 256, 0, stream>>>(h16[r], wh16[r], z, nullptr, nullptr,
                                           HID, GATE, 1);
      lstm_cell<<<(NB * HID) / 256, 256, 0, stream>>>(
          z, W[r][2], W[r][3], c32[r], h32[r], h16[r],
          enc + (size_t)l * NB * 1024 + (rev ? HID : 0), 1024);
    }
  };
  run_l1(2, wi16_e1f, false);
  run_l1(3, wi16_e1b, true);

  // ---- state projections -> decoder initial states ----
  auto proj = [&](const float* a, const float* b, const _Float16* wp,
                  const float* bp, float* d32, _Float16* d16) {
    concat2<<<(NB * 1024) / 256, 256, 0, stream>>>(catbuf, a, b);
    wmma_gemm<<<gHid, 256, 0, stream>>>(catbuf, wp, proj_pre, nullptr, bp,
                                        1024, HID, 0);
    softmax512<<<NB, 256, 0, stream>>>(proj_pre, d32, d16);
  };
  proj(h32[0], h32[2], hp1_16, hp1b, dh32[0], dh16[0]);   // h[0] = ph1
  proj(h32[1], h32[3], hp2_16, hp2b, dh32[1], dh16[1]);   // h[1] = ph2
  proj(c32[0], c32[2], cp1_16, cp1b, dc[0], nullptr);     // c[0] = pc1
  proj(c32[1], c32[3], cp2_16, cp2b, dc[1], nullptr);     // c[1] = pc2

  // ---- key/value projections over all L*N rows ----
  const dim3 gKV(HID / 128, (LSEQ * NB) / 16);            // 4 x 2048
  wmma_gemm<<<gKV, 256, 0, stream>>>(enc, kp_16, nullptr, key, kpb, 1024, HID, 0);
  wmma_gemm<<<gKV, 256, 0, stream>>>(enc, vp_16, nullptr, val, vpb, 1024, HID, 0);

  // ---- decoder (teacher forcing) ----
  for (int t = 0; t < TDEC; ++t) {
    attn_kernel<<<NB, 256, 0, stream>>>(dh32[1], key, val, ctx);
    build_x<<<(NB * 1024) / 256, 256, 0, stream>>>(xbuf, emb, words, t, ctx);
    // layer 0
    wmma_gemm<<<gGate, 256, 0, stream>>>(xbuf, wi16_d0, z, nullptr, nullptr,
                                         1024, GATE, 0);
    wmma_gemm<<<gGate, 256, 0, stream>>>(dh16[0], wh16[4], z, nullptr, nullptr,
                                         HID, GATE, 1);
    lstm_cell<<<(NB * HID) / 256, 256, 0, stream>>>(
        z, W[4][2], W[4][3], dc[0], dh32[0], dh16[0], nullptr, 0);
    // layer 1
    wmma_gemm<<<gGate, 256, 0, stream>>>(dh16[0], wi16_d1, z, nullptr, nullptr,
                                         HID, GATE, 0);
    wmma_gemm<<<gGate, 256, 0, stream>>>(dh16[1], wh16[5], z, nullptr, nullptr,
                                         HID, GATE, 1);
    lstm_cell<<<(NB * HID) / 256, 256, 0, stream>>>(
        z, W[5][2], W[5][3], dc[1], dh32[1], dh16[1], nullptr, 0);
    fc_softmax<<<NB, 32, 0, stream>>>(dh32[1], fcW, fcb, out, t);
  }
}
